// wide_basic_86182813761693
// MI455X (gfx1250) — matrix operand order repaired and checked by reading, not compiled, not run on hardware
//
#include <hip/hip_runtime.h>
#include <hip/hip_bf16.h>

// Problem constants (from reference)
#define B_    16
#define C_    320
#define HW_   1024      // 32*32
#define NH_   4
#define DK_   640
#define DV_   64
#define DKH_  160       // DK/NH
#define DVH_  16        // DV/NH
#define CO_   256
#define KTOT_ 2880      // C*9
#define NTOT_ 16384     // B*HW
#define QKVM_ 1344      // 2*DK+DV

typedef __bf16 bf16_t;
typedef bf16_t v16bf __attribute__((ext_vector_type(16)));
typedef float  v8f   __attribute__((ext_vector_type(8)));
typedef unsigned int v4u __attribute__((ext_vector_type(4)));
typedef int          v8i __attribute__((ext_vector_type(8)));
typedef int          v4i __attribute__((ext_vector_type(4)));

union FragU { unsigned int u[8]; uint4 q[2]; v16bf v; };

__device__ __forceinline__ v8f v8f_zero() {
  v8f z = {0.f, 0.f, 0.f, 0.f, 0.f, 0.f, 0.f, 0.f};
  return z;
}

// fp32 -> bf16 (round-to-nearest)
__device__ __forceinline__ unsigned short f2bf(float f) {
  union { float f; unsigned int u; } cv; cv.f = f;
  unsigned int u = cv.u;
  u += 0x7FFFu + ((u >> 16) & 1u);
  return (unsigned short)(u >> 16);
}

// ---------------------------------------------------------------------------
// 1) BN + ReLU -> bf16 activation, NCHW [B][C][1024]
// ---------------------------------------------------------------------------
__global__ void bnrelu_kernel(const float* __restrict__ in,
                              const float* __restrict__ g,  const float* __restrict__ bta,
                              const float* __restrict__ mu, const float* __restrict__ var,
                              unsigned short* __restrict__ act) {
  int idx = blockIdx.x * blockDim.x + threadIdx.x;
  if (idx >= B_ * C_ * HW_) return;
  int c = (idx >> 10) % C_;
  float sc = g[c] * rsqrtf(var[c] + 1e-5f);
  float y  = (in[idx] - mu[c]) * sc + bta[c];
  act[idx] = f2bf(fmaxf(y, 0.f));
}

// ---------------------------------------------------------------------------
// 2) weight fp32 -> bf16 (contiguous [M][C*9])
// ---------------------------------------------------------------------------
__global__ void w2bf_kernel(const float* __restrict__ w, unsigned short* __restrict__ o, int n) {
  int i = blockIdx.x * blockDim.x + threadIdx.x;
  if (i < n) o[i] = f2bf(w[i]);
}

// ---------------------------------------------------------------------------
// 3) 3x3 conv as implicit-GEMM WMMA:  D[m][n] = sum_k W[m][k] * im2col[k][n]
//    Block tile 128x128, BK=32, 8 waves (4Mx2N), wave tile 32x64 (2x4 frags).
//    A (weight) tile via TDM tensor_load_to_lds when available & tile is full.
//    LDS rows are 40 shorts (80 B) so fragments are 2 aligned b128 loads.
//    mode 0: dst [M][16384] (qkv);  mode 1: dst NCHW [B][320][1024]
// ---------------------------------------------------------------------------
__device__ __forceinline__ void loadA_coop(unsigned short (*lA)[40],
                                           const unsigned short* __restrict__ Wb,
                                           int m0, int k0, int Mreal, int tid) {
#pragma unroll
  for (int i = 0; i < 16; i++) {
    int idx = i * 256 + tid;
    int ml = idx >> 5, kl = idx & 31;   // k fastest across tid -> coalesced
    int m = m0 + ml; if (m >= Mreal) m = Mreal - 1;  // clamp (stores guarded)
    lA[ml][kl] = Wb[(size_t)m * KTOT_ + k0 + kl];
  }
}

__global__ __launch_bounds__(256)
void conv3x3_wmma_kernel(const unsigned short* __restrict__ Wb,   // bf16 [M][2880]
                         const unsigned short* __restrict__ Act,  // bf16 [B][320][1024]
                         const float* __restrict__ bias,
                         float* __restrict__ dst,
                         int Mreal, int mode, int chanOff) {
  __shared__ unsigned short lA[128][40];  // [m][k], 80B row (16B aligned)
  __shared__ unsigned short lB[128][40];  // [n][k] transposed for frag reads
  const int tid  = threadIdx.x;
  const int lane = tid & 31, wid = tid >> 5;
  const int wm = wid >> 1, wn = wid & 1;       // waves: 4 (M) x 2 (N)
  const int half = lane >> 4, lm = lane & 15;
  const int m0 = blockIdx.y * 128;
  const int n0 = blockIdx.x * 128;
  const bool fullTile = (m0 + 128) <= Mreal;
#if __has_builtin(__builtin_amdgcn_tensor_load_to_lds) && __has_builtin(__builtin_amdgcn_s_wait_tensorcnt)
  const int wuni = __builtin_amdgcn_readfirstlane((int)threadIdx.x) >> 5; // SGPR wave id
  const unsigned ldsA = (unsigned)(unsigned long long)(void*)&lA[0][0];
#endif

  v8f acc[2][4];
#pragma unroll
  for (int i = 0; i < 2; i++)
#pragma unroll
    for (int j = 0; j < 4; j++) acc[i][j] = v8f_zero();

  for (int k0 = 0; k0 < KTOT_; k0 += 32) {
    // ---- A (weights) tile: 128 x 32 bf16 ----
#if __has_builtin(__builtin_amdgcn_tensor_load_to_lds) && __has_builtin(__builtin_amdgcn_s_wait_tensorcnt)
    if (fullTile) {
      if (wuni == 0) {   // scalar branch: TDM ignores EXEC, so gate per-wave
        // D# group0: count=1 | lds_addr | global_addr(57b) | type=2
        unsigned long long ga =
            (unsigned long long)(const void*)(Wb + (size_t)m0 * KTOT_ + k0);
        v4u g0;
        g0[0] = 1u;
        g0[1] = ldsA;
        g0[2] = (unsigned)(ga & 0xFFFFFFFFu);
        g0[3] = (unsigned)((ga >> 32) & 0x01FFFFFFu) | (2u << 30);
        // D# group1: data_size=2B; LDS pad: every 16 DW insert 4 DW (64B->80B rows)
        v8i g1;
        g1[0] = (int)((1u << 16) | (1u << 20) | (3u << 22) | (3u << 25));
        g1[1] = (int)((1048576u & 0xFFFFu) << 16);          // tensor_dim0 lo (large: tile in-range)
        g1[2] = (int)((1048576u >> 16) | ((1048576u & 0xFFFFu) << 16)); // dim0 hi | dim1 lo
        g1[3] = (int)((1048576u >> 16) | (32u << 16));      // dim1 hi | tile_dim0 = 32 elems
        g1[4] = (int)(128u);                                // tile_dim1 = 128 rows
        g1[5] = (int)(KTOT_);                               // tensor_dim0_stride lo (elements)
        g1[6] = 0;
        g1[7] = 0;
        v4i z4 = {0, 0, 0, 0};
        v8i z8 = {0, 0, 0, 0, 0, 0, 0, 0};
        __builtin_amdgcn_tensor_load_to_lds(g0, g1, z4, z4, z8, 0);
        __builtin_amdgcn_s_wait_tensorcnt(0);
      }
    } else {
      loadA_coop(lA, Wb, m0, k0, Mreal, tid);
    }
#else
    loadA_coop(lA, Wb, m0, k0, Mreal, tid);
#endif
    // ---- B tile: implicit im2col, 128 pixels x 32 k (pixel-fastest: coalesced) ----
#pragma unroll
    for (int i = 0; i < 16; i++) {
      int idx = i * 256 + tid;
      int nl = idx & 127, kl = idx >> 7;
      int k = k0 + kl;
      int c  = k / 9;
      int r9 = k - c * 9;
      int kh = r9 / 3;
      int kw = r9 - kh * 3;
      int n = n0 + nl;
      int b = n >> 10, s = n & 1023;
      int h = (s >> 5) + kh - 1, w = (s & 31) + kw - 1;
      unsigned short v = 0;
      if (h >= 0 && h < 32 && w >= 0 && w < 32)
        v = Act[((size_t)(b * C_ + c) << 10) + (h << 5) + w];
      lB[nl][kl] = v;
    }
    __syncthreads();

    FragU af[2];
#pragma unroll
    for (int fm = 0; fm < 2; fm++) {   // A frag: K = 16*(j/4) + 8*half + 2*(j%4)
      const unsigned short* row = &lA[wm * 32 + fm * 16 + lm][0];
      af[fm].q[0] = *(const uint4*)&row[(half << 3)];
      af[fm].q[1] = *(const uint4*)&row[16 + (half << 3)];
    }
#pragma unroll
    for (int fn = 0; fn < 4; fn++) {
      FragU bf;                        // B frag: K = 16*(e/8) + 8*half + e%8, N = lm
      const unsigned short* row = &lB[wn * 64 + fn * 16 + lm][0];
      bf.q[0] = *(const uint4*)&row[(half << 3)];
      bf.q[1] = *(const uint4*)&row[16 + (half << 3)];
#pragma unroll
      for (int fm = 0; fm < 2; fm++)
        acc[fm][fn] = __builtin_amdgcn_wmma_f32_16x16x32_bf16(
            false, af[fm].v, false, bf.v, (short)0, acc[fm][fn], false, false);
    }
    __syncthreads();
  }

  // epilogue: C layout lane L: N = lm, M = r + 8*half
#pragma unroll
  for (int fm = 0; fm < 2; fm++) {
#pragma unroll
    for (int fn = 0; fn < 4; fn++) {
#pragma unroll
      for (int r = 0; r < 8; r++) {
        int m = m0 + wm * 32 + fm * 16 + r + (half << 3);
        int n = n0 + wn * 64 + fn * 16 + lm;
        if (m < Mreal) {
          float v = acc[fm][fn][r] + bias[m];
          if (mode == 0) {
            dst[(size_t)m * NTOT_ + n] = v;
          } else {
            int b = n >> 10, s = n & 1023;
            dst[((size_t)(b * C_ + chanOff + m) << 10) + s] = v;
          }
        }
      }
    }
  }
}

// ---------------------------------------------------------------------------
// 4) Relative-position bias tables:
//    RW[b][n][p][w2] = scale * dot(q[b,n,:,p], rel_w[w2 - w1 + 31])
//    RH[b][n][p][h2] = scale * dot(q[b,n,:,p], rel_h[h2 - h1 + 31])
//    (exact decomposition of _relative_logits_1d + _rel_to_abs)
// ---------------------------------------------------------------------------
__global__ void relbias_kernel(const float* __restrict__ qkv,   // [1344][16384]
                               const float* __restrict__ relw,  // [63][160]
                               const float* __restrict__ relh,  // [63][160]
                               float* __restrict__ RW, float* __restrict__ RH) {
  int idx = blockIdx.x * blockDim.x + threadIdx.x;   // 2^22 threads
  int m2 = idx & 63;
  int p  = (idx >> 6) & 1023;
  int n  = (idx >> 16) & 3;
  int b  = idx >> 18;
  const float qs = 0.07905694150420949f;  // 160^-0.5
  const float* rel;
  int ridx;
  if (m2 < 32) { ridx = m2 - (p & 31) + 31;        rel = relw; }
  else         { ridx = (m2 - 32) - (p >> 5) + 31; rel = relh; }
  float sum = 0.f;
  const float* qp = qkv + (size_t)(n * DKH_) * NTOT_ + b * HW_ + p;
  const float* rp = rel + ridx * DKH_;
  for (int f = 0; f < DKH_; f++) sum += qp[(size_t)f * NTOT_] * rp[f];
  sum *= qs;
  size_t o = ((size_t)((b * NH_ + n) * HW_ + p) << 5) + (m2 & 31);
  if (m2 < 32) RW[o] = sum; else RH[o] = sum;
}

// ---------------------------------------------------------------------------
// 5) Fused flash attention (WMMA): per block = (64-query tile, head, batch),
//    4 waves x 16 query rows. S = QK^T (K=160) + bias, online softmax, P*V.
//    LDS row strides are multiples of 8 shorts so frags are aligned b128 reads.
// ---------------------------------------------------------------------------
__global__ __launch_bounds__(128)
void attn_flash_kernel(const float* __restrict__ qkv,  // [1344][16384]
                       const float* __restrict__ RW, const float* __restrict__ RH,
                       float* __restrict__ attnbuf) {  // [B][64][1024]
  __shared__ unsigned short lQ[64][168];
  __shared__ unsigned short lK[64][168];
  __shared__ unsigned short lVt[16][72];   // [dvh][key]
  __shared__ unsigned short lP[64][72];    // [row][key]
  const int tid  = threadIdx.x;
  const int lane = tid & 31, wid = tid >> 5;
  const int half = lane >> 4, lm = lane & 15;
  const int q0 = blockIdx.x * 64;
  const int n  = blockIdx.y;
  const int b  = blockIdx.z;
  const float qs  = 0.07905694150420949f;
  const float L2E = 1.4426950408889634f;

  // load & scale Q tile (64 queries x 160 feats) -> bf16 LDS
  for (int i = 0; i < 80; i++) {
    int idx = i * 128 + tid;
    int pl = idx & 63, f = idx >> 6;
    float v = qkv[(size_t)(n * DKH_ + f) * NTOT_ + b * HW_ + q0 + pl] * qs;
    lQ[pl][f] = f2bf(v);
  }
  __syncthreads();

  FragU qa[5];  // Q A-fragments, resident for whole key loop
#pragma unroll
  for (int ks = 0; ks < 5; ks++) {
    const unsigned short* row = &lQ[wid * 16 + lm][0];
    qa[ks].q[0] = *(const uint4*)&row[ks * 32 + (half << 3)];
    qa[ks].q[1] = *(const uint4*)&row[ks * 32 + 16 + (half << 3)];
  }

  float mi[8], li[8];
#pragma unroll
  for (int r = 0; r < 8; r++) { mi[r] = -1e30f; li[r] = 0.f; }
  v8f oacc = v8f_zero();

  const int prowBase = (b * NH_ + n) * HW_ + q0 + wid * 16;

  for (int kt = 0; kt < 16; kt++) {
    int key0 = kt * 64;
    // K tile (64 keys x 160 feats)
    for (int i = 0; i < 80; i++) {
      int idx = i * 128 + tid;
      int kl = idx & 63, f = idx >> 6;
      lK[kl][f] = f2bf(qkv[(size_t)(DK_ + n * DKH_ + f) * NTOT_ + b * HW_ + key0 + kl]);
    }
    // V tile transposed (16 dvh x 64 keys)
    for (int i = 0; i < 8; i++) {
      int idx = i * 128 + tid;
      int kl = idx & 63, d = idx >> 6;
      lVt[d][kl] = f2bf(qkv[(size_t)(2 * DK_ + n * DVH_ + d) * NTOT_ + b * HW_ + key0 + kl]);
    }
    __syncthreads();

    // S = Q * K^T  (16 rows x 64 keys per wave)
    v8f s[4];
#pragma unroll
    for (int fn = 0; fn < 4; fn++) s[fn] = v8f_zero();
#pragma unroll
    for (int ks = 0; ks < 5; ks++) {
#pragma unroll
      for (int fn = 0; fn < 4; fn++) {
        FragU kb;
        const unsigned short* row = &lK[fn * 16 + lm][0];
        kb.q[0] = *(const uint4*)&row[ks * 32 + (half << 3)];
        kb.q[1] = *(const uint4*)&row[ks * 32 + 16 + (half << 3)];
        s[fn] = __builtin_amdgcn_wmma_f32_16x16x32_bf16(
            false, qa[ks].v, false, kb.v, (short)0, s[fn], false, false);
      }
    }

    // + relative bias, then online softmax per query row
#pragma unroll
    for (int r = 0; r < 8; r++) {
      size_t pb = ((size_t)(prowBase + r + (half << 3))) << 5;
#pragma unroll
      for (int fn = 0; fn < 4; fn++) {
        int qi = key0 + fn * 16 + lm;
        s[fn][r] += RW[pb + (qi & 31)] + RH[pb + (qi >> 5)];
      }
      float mx = fmaxf(fmaxf(s[0][r], s[1][r]), fmaxf(s[2][r], s[3][r]));
      mx = fmaxf(mx, __shfl_xor(mx, 1, 32));
      mx = fmaxf(mx, __shfl_xor(mx, 2, 32));
      mx = fmaxf(mx, __shfl_xor(mx, 4, 32));
      mx = fmaxf(mx, __shfl_xor(mx, 8, 32));
      float mnew = fmaxf(mi[r], mx);
      float corr = exp2f((mi[r] - mnew) * L2E);
      float rs = 0.f;
#pragma unroll
      for (int fn = 0; fn < 4; fn++) {
        float pv = exp2f((s[fn][r] - mnew) * L2E);
        s[fn][r] = pv;
        rs += pv;
      }
      rs += __shfl_xor(rs, 1, 32);
      rs += __shfl_xor(rs, 2, 32);
      rs += __shfl_xor(rs, 4, 32);
      rs += __shfl_xor(rs, 8, 32);
      li[r] = li[r] * corr + rs;
      oacc[r] *= corr;
      mi[r] = mnew;
      int row = wid * 16 + r + (half << 3);
#pragma unroll
      for (int fn = 0; fn < 4; fn++)
        lP[row][fn * 16 + lm] = f2bf(s[fn][r]);
    }
    __syncthreads();

    // O += P * V   (K = 64 keys -> 2 wmma steps)
#pragma unroll
    for (int ks = 0; ks < 2; ks++) {
      FragU pa, vb;
      const unsigned short* prow = &lP[wid * 16 + lm][0];
      pa.q[0] = *(const uint4*)&prow[ks * 32 + (half << 3)];
      pa.q[1] = *(const uint4*)&prow[ks * 32 + 16 + (half << 3)];
      const unsigned short* vrow = &lVt[lm][0];
      vb.q[0] = *(const uint4*)&vrow[ks * 32 + (half << 3)];
      vb.q[1] = *(const uint4*)&vrow[ks * 32 + 16 + (half << 3)];
      oacc = __builtin_amdgcn_wmma_f32_16x16x32_bf16(
          false, pa.v, false, vb.v, (short)0, oacc, false, false);
    }
    __syncthreads();
  }

  // normalize and scatter through raw reshape (b, NH, HW, dvh) -> (b, 64, 1024)
#pragma unroll
  for (int r = 0; r < 8; r++) {
    int p = q0 + wid * 16 + r + (half << 3);
    float v = oacc[r] / li[r];
    int t  = p * 16 + lm;          // flat within (HW, dvh)
    int ch = n * 16 + (t >> 10);
    attnbuf[((size_t)(b * DV_ + ch) << 10) + (t & 1023)] = v;
  }
}

// ---------------------------------------------------------------------------
// 6) 1x1 conv over attention output (64->64), write channels 256..319
// ---------------------------------------------------------------------------
__global__ void attn1x1_kernel(const float* __restrict__ attnbuf,  // [B][64][1024]
                               const float* __restrict__ w,        // [64][64]
                               const float* __restrict__ bias,
                               float* __restrict__ dst) {          // [B][320][1024]
  int idx = blockIdx.x * blockDim.x + threadIdx.x;  // 2^20
  int s  = idx & 1023;
  int co = (idx >> 10) & 63;
  int b  = idx >> 16;
  float sum = bias[co];
  const float* wr = w + co * DV_;
  const float* ab = attnbuf + ((size_t)(b * DV_) << 10) + s;
  for (int ci = 0; ci < DV_; ci++) sum += wr[ci] * ab[(size_t)ci << 10];
  dst[((size_t)(b * C_ + CO_ + co) << 10) + s] = sum;
}

// ---------------------------------------------------------------------------
// 7) residual add
// ---------------------------------------------------------------------------
__global__ void resadd_kernel(float* __restrict__ out, const float* __restrict__ x, int n) {
  int i = blockIdx.x * blockDim.x + threadIdx.x;
  if (i < n) out[i] += x[i];
}

// ---------------------------------------------------------------------------
extern "C" void kernel_launch(void* const* d_in, const int* in_sizes, int n_in,
                              void* d_out, int out_size, void* d_ws, size_t ws_size,
                              hipStream_t stream) {
  (void)in_sizes; (void)n_in; (void)ws_size;
  // setup_inputs() dict order:
  // 0:x 1-4:bn1_{g,b,m,v} 5-8:bn2_{g,b,m,v}
  // 9..16:  c1_{conv_w,conv_b,qkv_w,qkv_b,attn_w,attn_b,rel_w,rel_h}
  // 17..24: c2_{...}
  const float* x = (const float*)d_in[0];
  const float* bn_g[2] = {(const float*)d_in[1], (const float*)d_in[5]};
  const float* bn_b[2] = {(const float*)d_in[2], (const float*)d_in[6]};
  const float* bn_m[2] = {(const float*)d_in[3], (const float*)d_in[7]};
  const float* bn_v[2] = {(const float*)d_in[4], (const float*)d_in[8]};
  const float* conv_w[2] = {(const float*)d_in[9],  (const float*)d_in[17]};
  const float* conv_b[2] = {(const float*)d_in[10], (const float*)d_in[18]};
  const float* qkv_w[2]  = {(const float*)d_in[11], (const float*)d_in[19]};
  const float* qkv_b[2]  = {(const float*)d_in[12], (const float*)d_in[20]};
  const float* attn_w[2] = {(const float*)d_in[13], (const float*)d_in[21]};
  const float* attn_b[2] = {(const float*)d_in[14], (const float*)d_in[22]};
  const float* rel_w[2]  = {(const float*)d_in[15], (const float*)d_in[23]};
  const float* rel_h[2]  = {(const float*)d_in[16], (const float*)d_in[24]};

  char* ws = (char*)d_ws;
  size_t off = 0;
  auto carve = [&](size_t bytes) -> void* {
    void* p = ws + off;
    off += (bytes + 255) & ~(size_t)255;
    return p;
  };
  unsigned short* actbf = (unsigned short*)carve((size_t)B_ * C_ * HW_ * 2);     // 10.5 MB
  unsigned short* wbuf  = (unsigned short*)carve((size_t)QKVM_ * KTOT_ * 2);     // 7.7 MB
  float* qkvbuf = (float*)carve((size_t)QKVM_ * NTOT_ * 4);                      // 88 MB
  float* layer1 = (float*)carve((size_t)B_ * C_ * HW_ * 4);                      // 21 MB
  float* attnbf = (float*)carve((size_t)B_ * DV_ * HW_ * 4);                     // 4.2 MB
  float* RW     = (float*)carve((size_t)B_ * NH_ * HW_ * 32 * 4);                // 8.4 MB
  float* RH     = (float*)carve((size_t)B_ * NH_ * HW_ * 32 * 4);                // 8.4 MB

  for (int layer = 0; layer < 2; layer++) {
    const float* in = (layer == 0) ? x : layer1;
    float* outb = (layer == 0) ? layer1 : (float*)d_out;

    bnrelu_kernel<<<(B_ * C_ * HW_ + 255) / 256, 256, 0, stream>>>(
        in, bn_g[layer], bn_b[layer], bn_m[layer], bn_v[layer], actbf);

    // conv_out (256 ch) -> channels [0,256) of layer output
    w2bf_kernel<<<(CO_ * KTOT_ + 255) / 256, 256, 0, stream>>>(conv_w[layer], wbuf, CO_ * KTOT_);
    conv3x3_wmma_kernel<<<dim3(NTOT_ / 128, (CO_ + 127) / 128), 256, 0, stream>>>(
        wbuf, actbf, conv_b[layer], outb, CO_, /*mode=*/1, /*chanOff=*/0);

    // qkv conv (1344 ch) -> qkvbuf [1344][16384]
    w2bf_kernel<<<(QKVM_ * KTOT_ + 255) / 256, 256, 0, stream>>>(qkv_w[layer], wbuf, QKVM_ * KTOT_);
    conv3x3_wmma_kernel<<<dim3(NTOT_ / 128, (QKVM_ + 127) / 128), 256, 0, stream>>>(
        wbuf, actbf, qkv_b[layer], qkvbuf, QKVM_, /*mode=*/0, /*chanOff=*/0);

    // relative-position bias tables
    relbias_kernel<<<(B_ * NH_ * HW_ * 64) / 256, 256, 0, stream>>>(
        qkvbuf, rel_w[layer], rel_h[layer], RW, RH);

    // fused attention
    attn_flash_kernel<<<dim3(HW_ / 64, NH_, B_), 128, 0, stream>>>(qkvbuf, RW, RH, attnbf);

    // 1x1 conv -> channels [256,320) of layer output
    attn1x1_kernel<<<(B_ * DV_ * HW_) / 256, 256, 0, stream>>>(
        attnbf, attn_w[layer], attn_b[layer], outb);
  }

  resadd_kernel<<<(out_size + 255) / 256, 256, 0, stream>>>((float*)d_out, x, out_size);
}
